// GraphAttentionHead2_39496519254174
// MI455X (gfx1250) — matrix operand order repaired and checked by reading, not compiled, not run on hardware
//
#include <hip/hip_runtime.h>
#include <hip/hip_bf16.h>

#define B_    8
#define N_    2048
#define FIN   256
#define FOUT  64
#define ALPHA_ 0.2f
#define NEGINF_ (-9.0e15f)

typedef float  v8f   __attribute__((ext_vector_type(8)));
typedef float  v2f   __attribute__((ext_vector_type(2)));
typedef __bf16 v16bf __attribute__((ext_vector_type(16)));
typedef unsigned int v8u   __attribute__((ext_vector_type(8)));
typedef unsigned int u32x4 __attribute__((ext_vector_type(4)));
typedef int          i32x4 __attribute__((ext_vector_type(4)));
typedef int          i32x8 __attribute__((ext_vector_type(8)));

// ---------------------------------------------------------------------------
// Kernel 1: h = x @ W + b  via V_WMMA_F32_16X16X4_F32 (fp32 WMMA).
// One wave handles a 16-row tile (rows flat over B*N), producing 16x64.
// Epilogue: bias, bf16 transposed store hT[b][n][j], fused src/dst dots.
// ---------------------------------------------------------------------------
__global__ __launch_bounds__(128) void gat_h_kernel(
    const float* __restrict__ x, const float* __restrict__ W,
    const float* __restrict__ bW, const float* __restrict__ a_src,
    const float* __restrict__ a_dst, __bf16* __restrict__ hT,
    float* __restrict__ srcv, float* __restrict__ dstv)
{
  __shared__ float Wl[FIN * FOUT];
  for (int i = threadIdx.x; i < (FIN * FOUT) / 4; i += 128)
    ((float4*)Wl)[i] = ((const float4*)W)[i];
  __syncthreads();

  const int lane = threadIdx.x & 31;
  const int hf   = lane >> 4;        // which 16-lane half
  const int m    = lane & 15;
  const int ib   = (blockIdx.x * 4 + (threadIdx.x >> 5)) * 16;  // flat row base

  v8f c[4] = {};
  const float* xr = x + (size_t)(ib + m) * FIN;
  for (int k0 = 0; k0 < FIN; k0 += 4) {
    const int kb = k0 + hf * 2;                  // f32 A layout: lanes0-15 K0,K1; 16-31 K2,K3
    const v2f a = *(const v2f*)(xr + kb);
#pragma unroll
    for (int t = 0; t < 4; ++t) {
      const int n = t * 16 + m;
      v2f bv;
      bv[0] = Wl[kb * FOUT + n];
      bv[1] = Wl[(kb + 1) * FOUT + n];
      c[t] = __builtin_amdgcn_wmma_f32_16x16x4_f32(false, a, false, bv,
                                                   (short)0, c[t], false, false);
    }
  }

  float srow[8], drow[8];
#pragma unroll
  for (int r = 0; r < 8; ++r) { srow[r] = 0.f; drow[r] = 0.f; }

#pragma unroll
  for (int t = 0; t < 4; ++t) {
    const int n = t * 16 + m;
    const float bwv = bW[n], as = a_src[n], ad = a_dst[n];
#pragma unroll
    for (int r = 0; r < 8; ++r) {
      const float v = c[t][r] + bwv;             // C layout: row = r + hf*8, col = n
      const int g    = ib + r + hf * 8;
      const int bidx = g >> 11;
      const int irow = g & (N_ - 1);
      hT[((size_t)bidx * FOUT + n) * N_ + irow] = (__bf16)v;  // transposed bf16
      srow[r] += v * as;
      drow[r] += v * ad;
    }
  }
  // reduce the 16 lanes of each half (xor masks < 16 stay within the half)
#pragma unroll
  for (int r = 0; r < 8; ++r) {
#pragma unroll
    for (int off = 8; off >= 1; off >>= 1) {
      srow[r] += __shfl_xor(srow[r], off, 32);
      drow[r] += __shfl_xor(drow[r], off, 32);
    }
  }
  if (m == 0) {
#pragma unroll
    for (int r = 0; r < 8; ++r) {
      srcv[ib + hf * 8 + r] = srow[r];
      dstv[ib + hf * 8 + r] = drow[r];
    }
  }
}

// ---------------------------------------------------------------------------
// Kernel 2: per-row online softmax statistics (max, sum-exp).
// One wave per (b, i) row; the single HBM pass over adj (134 MB).
// ---------------------------------------------------------------------------
__global__ __launch_bounds__(256) void gat_rowstat_kernel(
    const int* __restrict__ adj, const float* __restrict__ srcv,
    const float* __restrict__ dstv, const float* __restrict__ a_b,
    float* __restrict__ rmax, float* __restrict__ rsum)
{
  const int wid  = (blockIdx.x * 256 + threadIdx.x) >> 5;  // 0..B*N-1
  const int lane = threadIdx.x & 31;
  const int b = wid >> 11;
  const float sv = srcv[wid] + a_b[0];
  const float* dr = dstv + b * N_;
  const int* ar   = adj + (size_t)wid * N_;

  float mmax = -3.0e38f, ssum = 0.f;
  for (int t = lane; t < N_; t += 32) {
    float e = sv + dr[t];
    e = e > 0.f ? e : ALPHA_ * e;
    e = ar[t] > 0 ? e : NEGINF_;
    if (e > mmax) { ssum *= __expf(mmax - e); mmax = e; }
    ssum += __expf(e - mmax);
  }
#pragma unroll
  for (int off = 16; off >= 1; off >>= 1) {
    const float om = __shfl_xor(mmax, off, 32);
    const float os = __shfl_xor(ssum, off, 32);
    const float M  = fmaxf(mmax, om);
    ssum = ssum * __expf(mmax - M) + os * __expf(om - M);
    mmax = M;
  }
  if (lane == 0) { rmax[wid] = mmax; rsum[wid] = ssum; }
}

// ---------------------------------------------------------------------------
// Kernel 3: out = elu(softmax(e) @ h) via V_WMMA_F32_16X16X32_BF16.
// Block = 8 waves = 128 rows of one batch. TDM stages hT[b] (64x2048 bf16,
// 256KB) into LDS once; adj re-read hits L2 (134MB < 192MB L2).
// ---------------------------------------------------------------------------
#define K2B_SMEM (FOUT * N_ * 2 + N_ * 4)   // 256KB hT + 8KB dst = 270336 B

__global__ __launch_bounds__(256) void gat_attn_kernel(
    const int* __restrict__ adj, const __bf16* __restrict__ hT,
    const float* __restrict__ srcv, const float* __restrict__ dstv,
    const float* __restrict__ a_b, const float* __restrict__ rmax,
    const float* __restrict__ rsum, float* __restrict__ out)
{
  extern __shared__ char smem[];
  __bf16* hl = (__bf16*)smem;                    // [FOUT][N_] (h transposed)
  float*  dl = (float*)(smem + FOUT * N_ * 2);   // [N_]

  const int b      = blockIdx.x >> 4;
  const int rowblk = (blockIdx.x & 15) * 128;

  for (int j = threadIdx.x; j < N_; j += 256) dl[j] = dstv[b * N_ + j];

  if (threadIdx.x < 32) {   // wave 0 issues the tensor DMA
    const unsigned long long ga =
        (unsigned long long)(const void*)(hT + (size_t)b * FOUT * N_);
    const unsigned la = (unsigned)(unsigned long long)(void*)hl;
    u32x4 g0;
    g0[0] = 1u;                                           // count=1 user desc
    g0[1] = la;                                           // lds_addr
    g0[2] = (unsigned)(ga & 0xFFFFFFFFull);               // global_addr[31:0]
    g0[3] = (unsigned)((ga >> 32) & 0x1FFFFFFull) | (2u << 30); // addr[56:32], type=2
    i32x8 g1;
    g1[0] = 1 << 16;                                      // data_size=2B
    g1[1] = (N_ & 0xFFFF) << 16;                          // tensor_dim0 lo
    g1[2] = ((N_ >> 16) & 0xFFFF) | ((FOUT & 0xFFFF) << 16); // dim0 hi | dim1 lo
    g1[3] = ((FOUT >> 16) & 0xFFFF) | ((N_ & 0xFFFF) << 16); // dim1 hi | tile_dim0
    g1[4] = FOUT & 0xFFFF;                                // tile_dim1 (tile_dim2=0)
    g1[5] = N_;                                           // tensor_dim0_stride lo32
    g1[6] = 0;
    g1[7] = 0;
    i32x4 z4 = {0, 0, 0, 0};
#if __clang_major__ >= 23
    i32x8 z8 = {0, 0, 0, 0, 0, 0, 0, 0};
    __builtin_amdgcn_tensor_load_to_lds(g0, g1, z4, z4, z8, 0);
#else
    __builtin_amdgcn_tensor_load_to_lds(g0, g1, z4, z4, 0);
#endif
    __builtin_amdgcn_s_wait_tensorcnt(0);
  }
  __syncthreads();

  const int w    = threadIdx.x >> 5;
  const int lane = threadIdx.x & 31;
  const int hf   = lane >> 4;
  const int m    = lane & 15;
  const int i0   = rowblk + w * 16;
  const int row  = i0 + m;                 // A-fragment row for this lane
  const size_t gr = (size_t)b * N_ + row;
  const float sv  = srcv[gr] + a_b[0];
  const float mm  = rmax[gr];
  const float inv = 1.0f / rsum[gr];
  const int* ar   = adj + gr * N_;

  v8f acc[4] = {};
  for (int j0 = 0; j0 < N_; j0 += 32) {
    // A fragment: normalized probabilities, 16-bit A layout
    // lanes0-15: V0..3 = K0..7, V4..7 = K16..23 ; lanes16-31: +8
    v8u araw;
#pragma unroll
    for (int v = 0; v < 8; ++v) {
      const int klo = (v < 4 ? 2 * v : 16 + 2 * (v - 4)) + hf * 8;
      const int j   = j0 + klo;
      const int2   av = *(const int2*)(ar + j);
      const float2 dv = *(const float2*)(dl + j);
      float e0 = sv + dv.x; e0 = e0 > 0.f ? e0 : ALPHA_ * e0; e0 = av.x > 0 ? e0 : NEGINF_;
      float e1 = sv + dv.y; e1 = e1 > 0.f ? e1 : ALPHA_ * e1; e1 = av.y > 0 ? e1 : NEGINF_;
      const float p0 = __expf(e0 - mm) * inv;
      const float p1 = __expf(e1 - mm) * inv;
      const unsigned lo = (unsigned)__builtin_bit_cast(unsigned short, (__bf16)p0);
      const unsigned hi = (unsigned)__builtin_bit_cast(unsigned short, (__bf16)p1);
      araw[v] = lo | (hi << 16);
    }
    const v16bf afrag = __builtin_bit_cast(v16bf, araw);

    // B fragments from LDS hT: lane holds column n; VGPR v = K pair (kb+o, kb+o+1), o = v<4 ? 2v : 8+2v
    const int kb = j0 + hf * 8;
#pragma unroll
    for (int t = 0; t < 4; ++t) {
      const int n = t * 16 + m;
      v8u braw;
#pragma unroll
      for (int v = 0; v < 8; ++v)
        braw[v] = *(const unsigned*)(hl + (size_t)n * N_ + kb + (v < 4 ? 2 * v : 8 + 2 * v));
      const v16bf bfrag = __builtin_bit_cast(v16bf, braw);
      acc[t] = __builtin_amdgcn_wmma_f32_16x16x32_bf16(false, afrag, false, bfrag,
                                                       (short)0, acc[t], false, false);
    }
  }

  // ELU epilogue + store
#pragma unroll
  for (int t = 0; t < 4; ++t) {
    const int n = t * 16 + m;
#pragma unroll
    for (int r = 0; r < 8; ++r) {
      float v = acc[t][r];
      v = v > 0.f ? v : __expf(v) - 1.f;
      const int mi = i0 + r + hf * 8;
      out[((size_t)b * N_ + mi) * FOUT + n] = v;
    }
  }
}

// ---------------------------------------------------------------------------
extern "C" void kernel_launch(void* const* d_in, const int* in_sizes, int n_in,
                              void* d_out, int out_size, void* d_ws, size_t ws_size,
                              hipStream_t stream) {
  const float* x     = (const float*)d_in[0];
  const int*   adj   = (const int*)d_in[1];
  const float* W     = (const float*)d_in[2];
  const float* bW    = (const float*)d_in[3];
  const float* a_src = (const float*)d_in[4];
  const float* a_dst = (const float*)d_in[5];
  const float* a_b   = (const float*)d_in[6];
  float* out = (float*)d_out;

  char* ws = (char*)d_ws;
  __bf16* hT  = (__bf16*)ws;                              // 2 MB bf16 h (transposed)
  float* srcv = (float*)(ws + (size_t)B_ * FOUT * N_ * 2);
  float* dstv = srcv + B_ * N_;
  float* rmax = dstv + B_ * N_;
  float* rsum = rmax + B_ * N_;

  gat_h_kernel<<<(B_ * N_) / 64, 128, 0, stream>>>(x, W, bW, a_src, a_dst,
                                                   hT, srcv, dstv);
  gat_rowstat_kernel<<<(B_ * N_) / 8, 256, 0, stream>>>(adj, srcv, dstv, a_b,
                                                        rmax, rsum);
  hipFuncSetAttribute((const void*)gat_attn_kernel,
                      hipFuncAttributeMaxDynamicSharedMemorySize, K2B_SMEM);
  gat_attn_kernel<<<B_ * (N_ / 128), 256, K2B_SMEM, stream>>>(
      adj, hT, srcv, dstv, a_b, rmax, rsum, out);
}
